// DecoderRNN_1881195676293
// MI455X (gfx1250) — compile-verified, matrix operand order checked by reading, not run on hardware
//
#include <hip/hip_runtime.h>
#include <hip/hip_bf16.h>

// Problem dimensions (fixed by the reference)
#define BB 128
#define NN 196
#define VD 512
#define HH 1024
#define EE 512
#define VV 10000
#define TT 40
#define G4H 4096          // 4*H
#define KCAT 2048         // 2E (=1024) + H (=1024)

typedef __bf16 bf16_t;
typedef __attribute__((ext_vector_type(16))) __bf16        v16bf;
typedef __attribute__((ext_vector_type(8)))  float         v8f;
typedef __attribute__((ext_vector_type(4)))  unsigned int  u32x4;

union FragBF {
    v16bf v;
    u32x4 u[2];
};

// ---------------------------------------------------------------------------
// Async global->LDS helpers (CDNA5, ASYNCcnt-tracked)
// VDST VGPR = per-lane LDS byte address (offset within workgroup LDS; the low
// 32 bits of a generic shared pointer), VADDR = 64-bit global address.
// ---------------------------------------------------------------------------
__device__ __forceinline__ void async_fill_b128(const bf16_t* gsrc, bf16_t* ldst) {
    unsigned lds_addr = (unsigned)(unsigned long long)(void*)ldst;
    unsigned long long gaddr = (unsigned long long)(const void*)gsrc;
    asm volatile("global_load_async_to_lds_b128 %0, %1, off"
                 :: "v"(lds_addr), "v"(gaddr)
                 : "memory");
}
__device__ __forceinline__ void wait_async_le1() {
    asm volatile("s_wait_asynccnt 0x1" ::: "memory");
}
__device__ __forceinline__ void wait_async_le0() {
    asm volatile("s_wait_asynccnt 0x0" ::: "memory");
}

// ---------------------------------------------------------------------------
// bf16 WMMA GEMM with LDS-staged B panel (NV = N-tiles per block, compile-time):
//   C[M,Ncols] = A[M,K] * W[Ncols,K]^T (+ bias[Ncols])
// Block = 256 threads (8 waves). Wave w owns M-tile (blockIdx.y*8 + w).
// Block covers N-tiles [nt_base + blockIdx.x*4, +NV) -> 64-row B panel in LDS,
// statically double-buffered per 32-wide K-chunk (K % 64 == 0 required),
// filled with global_load_async_to_lds_b128 (each of the 256 threads moves one
// 16-byte segment per chunk). All NV B fragments are loaded from LDS before
// the WMMA sequence so a single s_wait_dscnt covers NV back-to-back WMMAs.
//
// Fragment layouts per CDNA5 ISA (16-bit 16x32):
//   lane l, row = l&15, half h = l>>4:
//     elems [0..7]  = K = kk + 8h + 0..7
//     elems [8..15] = K = kk + 16 + 8h + 0..7
// C/D: lane l holds n = l&15, rows m = v + 8*(l>>4), v = 0..7.
// ---------------------------------------------------------------------------
template <int NV>
__global__ void __launch_bounds__(256)
wmma_gemm_bf16_lds(const bf16_t* __restrict__ A, int lda,
                   const bf16_t* __restrict__ Bw, int ldb,
                   float* __restrict__ C, long ldc,
                   const float* __restrict__ bias,
                   int nt_base, int Ntiles, int K)
{
    __shared__ __align__(16) bf16_t Bbuf0[64][32];
    __shared__ __align__(16) bf16_t Bbuf1[64][32];

    const int tid  = threadIdx.x;
    const int wave = tid >> 5;
    const int lane = tid & 31;
    const int nt0  = nt_base + blockIdx.x * 4;  // first N tile of this block
    const int mt   = blockIdx.y * 8 + wave;     // this wave's M tile
    const int r    = lane & 15;
    const int kh   = (lane >> 4) << 3;          // 0 or 8

    const int Ncols = Ntiles * 16;

    // B panel fill assignment: row = tid>>2 (0..63), segment = tid&3 (8 bf16 each)
    int wrow = nt0 * 16 + (tid >> 2);
    if (wrow >= Ncols) wrow = Ncols - 1;        // clamp rows beyond NV*16 (unused)
    const bf16_t* gB  = Bw + (size_t)wrow * (size_t)ldb + ((tid & 3) << 3);
    bf16_t* lB0 = &Bbuf0[tid >> 2][(tid & 3) << 3];
    bf16_t* lB1 = &Bbuf1[tid >> 2][(tid & 3) << 3];

    const bf16_t* arow = A + (size_t)(mt * 16 + r) * (size_t)lda;

    v8f acc[NV] = {};

    async_fill_b128(gB, lB0);                   // chunk 0 -> buf0
    async_fill_b128(gB + 32, lB1);              // chunk 1 -> buf1

    for (int kk = 0; kk < K; kk += 64) {
        // ---- chunk kk (buf0) ----
        wait_async_le1();                       // oldest fill (buf0) complete
        __syncthreads();
        {
            FragBF fa;
            fa.u[0] = *(const u32x4*)(arow + kk + kh);
            fa.u[1] = *(const u32x4*)(arow + kk + kh + 16);
            FragBF fb[NV];
#pragma unroll
            for (int j = 0; j < NV; ++j) {
                const bf16_t* bp = &Bbuf0[j * 16 + r][0];
                fb[j].u[0] = *(const u32x4*)(bp + kh);
                fb[j].u[1] = *(const u32x4*)(bp + kh + 16);
            }
#pragma unroll
            for (int j = 0; j < NV; ++j) {
                acc[j] = __builtin_amdgcn_wmma_f32_16x16x32_bf16(
                             false, fa.v, false, fb[j].v, (short)0, acc[j], false, false);
            }
        }
        __syncthreads();                        // panel reads done before refill
        const bool more = (kk + 64) < K;
        if (more) {
            async_fill_b128(gB + kk + 64, lB0); // next chunk for buf0
            __builtin_prefetch(arow + kk + 64 + kh, 0, 1);
            wait_async_le1();                   // buf1 fill complete
        } else {
            wait_async_le0();                   // drain: buf1 fill complete
        }
        // ---- chunk kk+32 (buf1) ----
        __syncthreads();
        {
            FragBF fa;
            fa.u[0] = *(const u32x4*)(arow + kk + 32 + kh);
            fa.u[1] = *(const u32x4*)(arow + kk + 32 + kh + 16);
            FragBF fb[NV];
#pragma unroll
            for (int j = 0; j < NV; ++j) {
                const bf16_t* bp = &Bbuf1[j * 16 + r][0];
                fb[j].u[0] = *(const u32x4*)(bp + kh);
                fb[j].u[1] = *(const u32x4*)(bp + kh + 16);
            }
#pragma unroll
            for (int j = 0; j < NV; ++j) {
                acc[j] = __builtin_amdgcn_wmma_f32_16x16x32_bf16(
                             false, fa.v, false, fb[j].v, (short)0, acc[j], false, false);
            }
        }
        __syncthreads();
        if ((kk + 96) < K) {
            async_fill_b128(gB + kk + 96, lB1); // next chunk for buf1
        }
    }

    const int n  = lane & 15;
    const int mb = (lane >> 4) << 3;
#pragma unroll
    for (int j = 0; j < NV; ++j) {
        const int ncol = (nt0 + j) * 16 + n;
        const float bv = bias ? bias[ncol] : 0.0f;
        float* crow = C + (size_t)(mt * 16 + mb) * (size_t)ldc + (size_t)ncol;
#pragma unroll
        for (int v = 0; v < 8; ++v)
            crow[(size_t)v * (size_t)ldc] = acc[j][v] + bv;
    }
}

// ---------------------------------------------------------------------------
// Setup / pointwise kernels
// ---------------------------------------------------------------------------
__global__ void f32_to_bf16_k(const float* __restrict__ src, bf16_t* __restrict__ dst, long n) {
    long i = (long)blockIdx.x * blockDim.x + threadIdx.x;
    if (i < n) dst[i] = (bf16_t)src[i];
}

__global__ void zero_f32_k(float* __restrict__ p, long n) {
    long i = (long)blockIdx.x * blockDim.x + threadIdx.x;
    if (i < n) p[i] = 0.0f;
}

__global__ void zero_bf16_k(bf16_t* __restrict__ p, long n) {
    long i = (long)blockIdx.x * blockDim.x + threadIdx.x;
    if (i < n) p[i] = (bf16_t)0.0f;
}

// Wcat[r, 0:1024] = W_ih[r,:], Wcat[r, 1024:2048] = W_hh[r,:]   (bf16)
__global__ void build_wcat_k(const float* __restrict__ W_ih, const float* __restrict__ W_hh,
                             bf16_t* __restrict__ Wcat) {
    long i = (long)blockIdx.x * blockDim.x + threadIdx.x;
    if (i >= (long)G4H * KCAT) return;
    int r = (int)(i >> 11);           // /2048
    int c = (int)(i & 2047);
    float v = (c < 1024) ? W_ih[(size_t)r * 1024 + c]
                         : W_hh[(size_t)r * 1024 + (c - 1024)];
    Wcat[i] = (bf16_t)v;
}

__global__ void bias_sum_k(const float* __restrict__ a, const float* __restrict__ b,
                           float* __restrict__ o) {
    int i = blockIdx.x * blockDim.x + threadIdx.x;
    if (i < G4H) o[i] = a[i] + b[i];
}

__global__ void feats_mean_k(const float* __restrict__ features, float* __restrict__ fm) {
    int i = blockIdx.x * blockDim.x + threadIdx.x;     // B*VD
    if (i >= BB * VD) return;
    int b = i / VD, d = i % VD;
    const float* p = features + (size_t)b * NN * VD + d;
    float s = 0.0f;
    for (int n = 0; n < NN; ++n) s += p[(size_t)n * VD];
    fm[i] = s * (1.0f / (float)NN);
}

// embs[b,t,:] = (t==0) ? feats_mean[b,:] : embed_table[captions[b,t-1],:]   (bf16)
__global__ void build_embs_k(const float* __restrict__ fm, const float* __restrict__ table,
                             const int* __restrict__ captions, bf16_t* __restrict__ embs) {
    long i = (long)blockIdx.x * blockDim.x + threadIdx.x;
    if (i >= (long)BB * TT * EE) return;
    int e = (int)(i % EE);
    long bt = i / EE;
    int t = (int)(bt % TT);
    int b = (int)(bt / TT);
    float v;
    if (t == 0) v = fm[(size_t)b * EE + e];
    else        v = table[(size_t)captions[(size_t)b * TT + (t - 1)] * EE + e];
    embs[i] = (bf16_t)v;
}

// ---------------------------------------------------------------------------
// Attention: scores = relu(att_fea + att_h + att_bias) . att_w ; softmax over N;
// context = sum_n alpha_n * features[b,n,:]  -> xcat[b,0:512] (bf16)
// also copies emb_t -> xcat[b,512:1024]
// One block (256 threads) per batch element.
// ---------------------------------------------------------------------------
__global__ void __launch_bounds__(256)
attention_context_k(const float* __restrict__ att_fea,   // [B,N,VD]
                    const float* __restrict__ att_h,     // [B,VD]
                    const float* __restrict__ att_bias,  // [N]
                    const float* __restrict__ att_w,     // [VD]
                    const float* __restrict__ features,  // [B,N,VD] (f32 original)
                    const bf16_t* __restrict__ embs,     // [B,T,E]
                    int t,
                    bf16_t* __restrict__ xcat)           // [B,KCAT]
{
    const int b = blockIdx.x;
    const int tid = threadIdx.x;

    __shared__ float sh_h[VD];
    __shared__ float sh_w[VD];
    __shared__ float sc[NN];
    __shared__ float red[256];

    for (int d = tid; d < VD; d += 256) {
        sh_h[d] = att_h[(size_t)b * VD + d];
        sh_w[d] = att_w[d];
    }
    __syncthreads();

    for (int n = tid; n < NN; n += 256) {
        const float* fr = att_fea + ((size_t)b * NN + n) * VD;
        const float bn = att_bias[n];
        float s = 0.0f;
        for (int d = 0; d < VD; ++d) {
            float v = fr[d] + sh_h[d] + bn;
            s += (v > 0.0f ? v : 0.0f) * sh_w[d];
        }
        sc[n] = s;
    }
    __syncthreads();

    // softmax max
    float m = -3.4e38f;
    for (int n = tid; n < NN; n += 256) m = fmaxf(m, sc[n]);
    red[tid] = m; __syncthreads();
    for (int s = 128; s > 0; s >>= 1) {
        if (tid < s) red[tid] = fmaxf(red[tid], red[tid + s]);
        __syncthreads();
    }
    m = red[0];
    __syncthreads();

    // softmax exp + sum
    float sum = 0.0f;
    for (int n = tid; n < NN; n += 256) {
        float e = __expf(sc[n] - m);
        sc[n] = e;
        sum += e;
    }
    red[tid] = sum; __syncthreads();
    for (int s = 128; s > 0; s >>= 1) {
        if (tid < s) red[tid] += red[tid + s];
        __syncthreads();
    }
    const float inv = 1.0f / red[0];

    // context over d
    for (int d = tid; d < VD; d += 256) {
        const float* fb = features + (size_t)b * NN * VD + d;
        float acc = 0.0f;
        for (int n = 0; n < NN; ++n) acc += sc[n] * fb[(size_t)n * VD];
        xcat[(size_t)b * KCAT + d] = (bf16_t)(acc * inv);
    }
    // copy embedding slice
    const bf16_t* eb = embs + (size_t)b * TT * EE + (size_t)t * EE;
    for (int e = tid; e < EE; e += 256) {
        xcat[(size_t)b * KCAT + VD + e] = eb[e];
    }
}

// ---------------------------------------------------------------------------
// LSTM pointwise: gates [B,4H] (i,f,g,o) ; updates cx (f32); writes hx (bf16)
// into hxbf and xcat[:,1024:2048].
// ---------------------------------------------------------------------------
__device__ __forceinline__ float sigmoidf(float x) { return 1.0f / (1.0f + __expf(-x)); }

__global__ void lstm_pointwise_k(const float* __restrict__ gates,
                                 float* __restrict__ cx,
                                 bf16_t* __restrict__ hxbf,
                                 bf16_t* __restrict__ xcat)
{
    int idx = blockIdx.x * blockDim.x + threadIdx.x;   // B*H
    if (idx >= BB * HH) return;
    int b = idx >> 10, h = idx & (HH - 1);
    const float* g = gates + (size_t)b * G4H;
    float gi = sigmoidf(g[h]);
    float gf = sigmoidf(g[HH + h]);
    float gg = tanhf(g[2 * HH + h]);
    float go = sigmoidf(g[3 * HH + h]);
    float c  = gf * cx[idx] + gi * gg;
    cx[idx]  = c;
    bf16_t hb = (bf16_t)(go * tanhf(c));
    hxbf[idx] = hb;
    xcat[(size_t)b * KCAT + 1024 + h] = hb;
}

// ---------------------------------------------------------------------------
// Host side
// ---------------------------------------------------------------------------
static inline int cdiv(long a, long b) { return (int)((a + b - 1) / b); }

// Requires Mtiles % 8 == 0 (true for all call sites: 1568, 8, 8, 8)
// and K % 64 == 0 (512, 1024, 2048).
static void launch_gemm(const bf16_t* A, int lda, const bf16_t* Bw, int ldb,
                        float* C, long ldc, const float* bias,
                        int Mtiles, int Ntiles, int K, hipStream_t stream)
{
    const int full = Ntiles >> 2;
    const int rem  = Ntiles & 3;
    const unsigned mgrid = (unsigned)(Mtiles / 8);
    if (full) {
        wmma_gemm_bf16_lds<4><<<dim3((unsigned)full, mgrid), 256, 0, stream>>>(
            A, lda, Bw, ldb, C, ldc, bias, 0, Ntiles, K);
    }
    if (rem == 1) {
        wmma_gemm_bf16_lds<1><<<dim3(1, mgrid), 256, 0, stream>>>(
            A, lda, Bw, ldb, C, ldc, bias, full * 4, Ntiles, K);
    } else if (rem == 2) {
        wmma_gemm_bf16_lds<2><<<dim3(1, mgrid), 256, 0, stream>>>(
            A, lda, Bw, ldb, C, ldc, bias, full * 4, Ntiles, K);
    } else if (rem == 3) {
        wmma_gemm_bf16_lds<3><<<dim3(1, mgrid), 256, 0, stream>>>(
            A, lda, Bw, ldb, C, ldc, bias, full * 4, Ntiles, K);
    }
}

extern "C" void kernel_launch(void* const* d_in, const int* in_sizes, int n_in,
                              void* d_out, int out_size, void* d_ws, size_t ws_size,
                              hipStream_t stream)
{
    const float* features    = (const float*)d_in[0];   // [B,N,VD]
    const int*   captions    = (const int*)  d_in[1];   // [B,T]
    // d_in[2] lengths: unused (matches reference)
    const float* embed_table = (const float*)d_in[3];   // [V,E]
    const float* W_ih        = (const float*)d_in[4];   // [4H,2E]
    const float* b_ih        = (const float*)d_in[5];   // [4H]
    const float* W_hh        = (const float*)d_in[6];   // [4H,H]
    const float* b_hh        = (const float*)d_in[7];   // [4H]
    const float* W_out       = (const float*)d_in[8];   // [V,H]
    const float* b_out       = (const float*)d_in[9];   // [V]
    const float* att_vw      = (const float*)d_in[10];  // [VD,VD]
    const float* att_hw      = (const float*)d_in[11];  // [VD,H]
    const float* att_bias    = (const float*)d_in[12];  // [N]
    const float* att_w       = (const float*)d_in[13];  // [1,VD]
    float* out = (float*)d_out;                         // [B,T,V]

    // ---- workspace carve (256B aligned) ----
    char* ws = (char*)d_ws;
    auto alloc = [&](size_t bytes) -> void* {
        void* p = (void*)ws;
        ws += (bytes + 255) & ~(size_t)255;
        return p;
    };
    bf16_t* feat_bf = (bf16_t*)alloc((size_t)BB * NN * VD * sizeof(bf16_t));
    float*  attfea  = (float*) alloc((size_t)BB * NN * VD * sizeof(float));
    bf16_t* vw_bf   = (bf16_t*)alloc((size_t)VD * VD * sizeof(bf16_t));
    bf16_t* hw_bf   = (bf16_t*)alloc((size_t)VD * HH * sizeof(bf16_t));
    bf16_t* wout_bf = (bf16_t*)alloc((size_t)VV * HH * sizeof(bf16_t));
    bf16_t* wcat_bf = (bf16_t*)alloc((size_t)G4H * KCAT * sizeof(bf16_t));
    float*  bias4   = (float*) alloc((size_t)G4H * sizeof(float));
    float*  fmean   = (float*) alloc((size_t)BB * EE * sizeof(float));
    bf16_t* embs    = (bf16_t*)alloc((size_t)BB * TT * EE * sizeof(bf16_t));
    bf16_t* hxbf    = (bf16_t*)alloc((size_t)BB * HH * sizeof(bf16_t));
    float*  cx      = (float*) alloc((size_t)BB * HH * sizeof(float));
    bf16_t* xcat    = (bf16_t*)alloc((size_t)BB * KCAT * sizeof(bf16_t));
    float*  atth    = (float*) alloc((size_t)BB * VD * sizeof(float));
    float*  gates   = (float*) alloc((size_t)BB * G4H * sizeof(float));

    // ---- one-time setup ----
    {
        long n;
        n = (long)BB * NN * VD;
        f32_to_bf16_k<<<cdiv(n, 256), 256, 0, stream>>>(features, feat_bf, n);
        n = (long)VD * VD;
        f32_to_bf16_k<<<cdiv(n, 256), 256, 0, stream>>>(att_vw, vw_bf, n);
        n = (long)VD * HH;
        f32_to_bf16_k<<<cdiv(n, 256), 256, 0, stream>>>(att_hw, hw_bf, n);
        n = (long)VV * HH;
        f32_to_bf16_k<<<cdiv(n, 256), 256, 0, stream>>>(W_out, wout_bf, n);
        n = (long)G4H * KCAT;
        build_wcat_k<<<cdiv(n, 256), 256, 0, stream>>>(W_ih, W_hh, wcat_bf);
        bias_sum_k<<<cdiv(G4H, 256), 256, 0, stream>>>(b_ih, b_hh, bias4);
        feats_mean_k<<<cdiv((long)BB * VD, 256), 256, 0, stream>>>(features, fmean);
        n = (long)BB * TT * EE;
        build_embs_k<<<cdiv(n, 256), 256, 0, stream>>>(fmean, embed_table, captions, embs);
        // zero state (harness poisons ws; we must init every call)
        zero_f32_k<<<cdiv((long)BB * HH, 256), 256, 0, stream>>>(cx, (long)BB * HH);
        zero_bf16_k<<<cdiv((long)BB * HH, 256), 256, 0, stream>>>(hxbf, (long)BB * HH);
        zero_bf16_k<<<cdiv((long)BB * KCAT, 256), 256, 0, stream>>>(xcat, (long)BB * KCAT);
    }

    // att_fea = features @ att_vw^T   -> f32 [B*N, VD]
    launch_gemm(feat_bf, VD, vw_bf, VD, attfea, VD, nullptr,
                (BB * NN) / 16, VD / 16, VD, stream);

    // ---- sequential decode over T steps ----
    for (int t = 0; t < TT; ++t) {
        // att_h = hx @ att_hw^T : [B,VD]
        launch_gemm(hxbf, HH, hw_bf, HH, atth, VD, nullptr,
                    BB / 16, VD / 16, HH, stream);

        // attention -> context (bf16) + emb copy into xcat[:, :1024]
        attention_context_k<<<BB, 256, 0, stream>>>(attfea, atth, att_bias, att_w,
                                                    features, embs, t, xcat);

        // gates = xcat @ Wcat^T + (b_ih + b_hh) : [B,4H]
        launch_gemm(xcat, KCAT, wcat_bf, KCAT, gates, G4H, bias4,
                    BB / 16, G4H / 16, KCAT, stream);

        // LSTM pointwise: cx update, hx -> bf16 (hxbf and xcat[:,1024:2048])
        lstm_pointwise_k<<<cdiv((long)BB * HH, 256), 256, 0, stream>>>(gates, cx, hxbf, xcat);

        // out[:, t, :] = hx @ W_out^T + b_out : [B,V], ldc = T*V
        launch_gemm(hxbf, HH, wout_bf, HH, out + (size_t)t * VV, (long)TT * VV, b_out,
                    BB / 16, VV / 16, HH, stream);
    }
}
